// LanguageModel_35184372089156
// MI455X (gfx1250) — matrix operand order repaired and checked by reading, not compiled, not run on hardware
//
#include <hip/hip_runtime.h>
#include <cstdint>
#include <cstddef>

// Model constants (fixed by the reference)
#define DMODEL 1024
#define NHEAD  16
#define HSZ    64
#define DFFN   4096
#define NLAYER 6
#define SEQ    128
#define BATCH  16
#define MROWS  (BATCH * SEQ)   // 2048
#define VOCAB  50257

typedef __attribute__((ext_vector_type(16))) _Float16 v16h;
typedef __attribute__((ext_vector_type(8)))  float    v8f;

union AccU  { v8f  v; float f[8]; };
union FragU { v16h v; uint4 u[2]; };
union H4U   { uint2 u2; _Float16 h[4]; };

// Async global->LDS copy of 16 bytes per lane (CDNA5, tracked by ASYNCcnt).
__device__ __forceinline__ void async_copy_b128(const void* gptr, void* lptr) {
    unsigned lds = (unsigned)(uintptr_t)lptr;   // low 32 bits = LDS offset
    asm volatile("global_load_async_to_lds_b128 %0, %1, off"
                 :: "v"(lds), "v"(gptr) : "memory");
}
__device__ __forceinline__ void async_wait0() {
    asm volatile("s_wait_asynccnt 0x0" ::: "memory");
}

// ---------------------------------------------------------------- embedding
__global__ __launch_bounds__(256) void embed_kernel(
    const int* __restrict__ idx, const float* __restrict__ tok,
    const float* __restrict__ pos, float* __restrict__ x) {
    int bt = blockIdx.x;                 // 0..2047
    int t  = bt & (SEQ - 1);
    int token = idx[bt];
    const float4* tr = (const float4*)(tok + (size_t)token * DMODEL);
    const float4* pr = (const float4*)(pos + (size_t)t * DMODEL);
    float4* xr = (float4*)(x + (size_t)bt * DMODEL);
    int i = threadIdx.x;                 // 256 threads x float4 = 1024
    float4 a = tr[i], b = pr[i];
    xr[i] = make_float4(a.x + b.x, a.y + b.y, a.z + b.z, a.w + b.w);
}

// ---------------------------------------------------------------- layernorm
// One row per block; each thread owns exactly one float4 (kept in registers
// across mean/var/normalize so x is read once).
__global__ __launch_bounds__(256) void layernorm_kernel(
    const float* __restrict__ x, const float* __restrict__ w,
    const float* __restrict__ b, float* __restrict__ out) {
    __shared__ float red[256];
    const int row = blockIdx.x, tid = threadIdx.x;
    const float4 xv = ((const float4*)(x + (size_t)row * DMODEL))[tid];
    float s = xv.x + xv.y + xv.z + xv.w;
    red[tid] = s; __syncthreads();
    for (int off = 128; off; off >>= 1) {
        if (tid < off) red[tid] += red[tid + off];
        __syncthreads();
    }
    const float mean = red[0] * (1.f / DMODEL);
    __syncthreads();
    float d0 = xv.x - mean, d1 = xv.y - mean, d2 = xv.z - mean, d3 = xv.w - mean;
    red[tid] = d0 * d0 + d1 * d1 + d2 * d2 + d3 * d3;
    __syncthreads();
    for (int off = 128; off; off >>= 1) {
        if (tid < off) red[tid] += red[tid + off];
        __syncthreads();
    }
    const float rstd = rsqrtf(red[0] * (1.f / DMODEL) + 1e-5f);
    const float4 wv = ((const float4*)w)[tid];
    const float4 bv = ((const float4*)b)[tid];
    float4 o = make_float4(d0 * rstd * wv.x + bv.x, d1 * rstd * wv.y + bv.y,
                           d2 * rstd * wv.z + bv.z, d3 * rstd * wv.w + bv.w);
    ((float4*)(out + (size_t)row * DMODEL))[tid] = o;
}

// ---------------------------------------------------------------- WMMA GEMM
// C[2048,N] = epi(A[2048,K] @ B[K,N] + bias [+ resid]) via
// v_wmma_f32_16x16x32_f16.  BMODE 0: B row-major [K,N].  BMODE 1: per-head
// weight W[H][K][64]; logical col j -> W[j>>6][k][j&63].  OUTF16 writes f16.
// 256 threads = 8 waves; 128x128 tile, BK=32; register-staged pipeline:
// prefetch tile k+1 (float4 global loads) while WMMAs consume tile k.
template<int BMODE, bool BIAS, bool RELU, bool RESID, bool OUTF16>
__global__ __launch_bounds__(256) void gemm_wmma_kernel(
    const float* __restrict__ A, const float* __restrict__ Bw,
    const float* __restrict__ bias, const float* __restrict__ resid,
    void* __restrict__ Cv, int N, int K) {
    __shared__ __align__(16) _Float16 As[128][40];   // [m][k], 80B rows
    __shared__ __align__(16) _Float16 Bs[128][40];   // [n][k] (transposed fill)
    const int tid  = threadIdx.x;
    const int lane = tid & 31;
    const int wave = tid >> 5;
    const int rowBlock = blockIdx.y << 7;
    const int colBlock = blockIdx.x << 7;
    // Vector path for B only when the full 128-wide tile is in range and rows
    // are 16B-aligned (head GEMM: N=50257 -> scalar guarded path).
    const bool vecB = (BMODE == 1) || (((N & 3) == 0) && (colBlock + 128 <= N));

    AccU acc[8];
    #pragma unroll
    for (int j = 0; j < 8; ++j)
        #pragma unroll
        for (int r = 0; r < 8; ++r) acc[j].f[r] = 0.f;

    float4 aReg[4], bReg[4];

    auto loadTiles = [&](int k0) {
        #pragma unroll
        for (int u = 0; u < 4; ++u) {              // A: 128x32 = 1024 float4
            int i = tid + (u << 8);
            int r = i >> 3, c4 = (i & 7) << 2;
            aReg[u] = *(const float4*)(A + (size_t)(rowBlock + r) * K + k0 + c4);
        }
        #pragma unroll
        for (int u = 0; u < 4; ++u) {              // B: 32x128 = 1024 float4
            int i = tid + (u << 8);
            int r = i >> 5, c4 = (i & 31) << 2;
            if (vecB) {
                const float* src;
                if (BMODE == 0) {
                    src = Bw + (size_t)(k0 + r) * N + colBlock + c4;
                } else {
                    int col = colBlock + c4;       // never crosses a head (c4%4==0)
                    src = Bw + ((size_t)(col >> 6) * K + (k0 + r)) * 64 + (col & 63);
                }
                bReg[u] = *(const float4*)src;
            } else {
                float e[4];
                #pragma unroll
                for (int q = 0; q < 4; ++q) {
                    int col = colBlock + c4 + q;
                    e[q] = (col < N) ? Bw[(size_t)(k0 + r) * N + col] : 0.f;
                }
                bReg[u] = make_float4(e[0], e[1], e[2], e[3]);
            }
        }
    };
    auto storeTiles = [&]() {
        #pragma unroll
        for (int u = 0; u < 4; ++u) {
            int i = tid + (u << 8);
            int r = i >> 3, c4 = (i & 7) << 2;
            H4U p;
            p.h[0] = (_Float16)aReg[u].x; p.h[1] = (_Float16)aReg[u].y;
            p.h[2] = (_Float16)aReg[u].z; p.h[3] = (_Float16)aReg[u].w;
            *(uint2*)&As[r][c4] = p.u2;            // b64 store
        }
        #pragma unroll
        for (int u = 0; u < 4; ++u) {
            int i = tid + (u << 8);
            int r = i >> 5, c4 = (i & 31) << 2;
            Bs[c4 + 0][r] = (_Float16)bReg[u].x;
            Bs[c4 + 1][r] = (_Float16)bReg[u].y;
            Bs[c4 + 2][r] = (_Float16)bReg[u].z;
            Bs[c4 + 3][r] = (_Float16)bReg[u].w;
        }
    };

    // CDNA5 16-bit A-frag (16x32): lane m=lane&15; lanes<16 carry K 0-7/16-23,
    // lanes>=16 carry K 8-15/24-31 -> two b128 LDS loads at k={base, base+16}.
    const int mrow   = lane & 15;
    const int aBaseK = (lane >> 4) << 3;   // 0 or 8
    // B-frag (32x16): col n=lane&15; lanes<16 hold K 0-7/16-23, lanes>=16 K 8-15/24-31.
    const int bBaseK = (lane >> 4) << 3;   // 0 or 8

    auto computeTile = [&]() {
        FragU fa;
        const _Float16* arow = &As[(wave << 4) + mrow][0];
        fa.u[0] = *(const uint4*)(arow + aBaseK);
        fa.u[1] = *(const uint4*)(arow + aBaseK + 16);
        #pragma unroll
        for (int j = 0; j < 8; ++j) {
            FragU fb;
            const _Float16* brow = &Bs[(j << 4) + (lane & 15)][0];
            fb.u[0] = *(const uint4*)(brow + bBaseK);
            fb.u[1] = *(const uint4*)(brow + bBaseK + 16);
            acc[j].v = __builtin_amdgcn_wmma_f32_16x16x32_f16(
                false, fa.v, false, fb.v, (short)0, acc[j].v, false, false);
        }
    };

    loadTiles(0);
    storeTiles();
    __syncthreads();
    for (int k0 = 32; k0 < K; k0 += 32) {
        loadTiles(k0);      // prefetch into registers (overlaps WMMA below)
        computeTile();      // tile k0-32
        __syncthreads();    // all waves done reading LDS
        storeTiles();
        __syncthreads();    // new tile visible
    }
    computeTile();          // last tile

    // C/D layout: VGPR r holds M=r (lanes 0-15) / M=r+8 (lanes 16-31), N=lane&15
    const int rOff = (lane >> 4) << 3;
    #pragma unroll
    for (int j = 0; j < 8; ++j) {
        int col = colBlock + (j << 4) + (lane & 15);
        if (col < N) {
            #pragma unroll
            for (int r = 0; r < 8; ++r) {
                int row = rowBlock + (wave << 4) + r + rOff;
                float val = acc[j].f[r];
                if (BIAS)  val += bias[col];
                if (RELU)  val = fmaxf(val, 0.f);
                if (RESID) val += resid[(size_t)row * N + col];
                if (OUTF16) ((_Float16*)Cv)[(size_t)row * N + col] = (_Float16)val;
                else        ((float*)Cv)[(size_t)row * N + col]    = val;
            }
        }
    }
}

// ---------------------------------------------------------------- attention
// q,k,v are f16 [B*T, D] with column = head*64 + c (written by the QKV GEMMs).
// One block per (b,h); K/V tiles pulled into LDS with async global->LDS B128
// copies (ASYNCcnt), per-wave row softmax (16 rows/wave), scale = D^-0.5.
__global__ __launch_bounds__(256) void attention_kernel(
    const _Float16* __restrict__ q, const _Float16* __restrict__ k,
    const _Float16* __restrict__ v, float* __restrict__ out) {
    __shared__ __align__(16) _Float16 Ks[SEQ][72];   // 144B rows (16B multiple)
    __shared__ __align__(16) _Float16 Vs[SEQ][72];
    __shared__ float Qrow[8][HSZ];
    __shared__ float Ps[8][SEQ];
    const int b = blockIdx.x >> 4;
    const int h = blockIdx.x & 15;
    const int tid = threadIdx.x, lane = tid & 31, wave = tid >> 5;
    const size_t base = ((size_t)b * SEQ) * DMODEL + (size_t)h * HSZ;

    // 128 rows x 64 halves = 1024 16B chunks per matrix; 4 per thread each.
    #pragma unroll
    for (int u = 0; u < 4; ++u) {
        int i = tid + (u << 8);
        int t = i >> 3, c8 = (i & 7) << 3;
        async_copy_b128(k + base + (size_t)t * DMODEL + c8, &Ks[t][c8]);
        async_copy_b128(v + base + (size_t)t * DMODEL + c8, &Vs[t][c8]);
    }
    async_wait0();
    __syncthreads();

    const float scale = 0.03125f;  // 1/sqrt(D)=1/32 (reference scales by n_embd)
    for (int i = 0; i < 16; ++i) {
        const int t = (wave << 4) + i;
        Qrow[wave][lane]      = (float)q[base + (size_t)t * DMODEL + lane];
        Qrow[wave][lane + 32] = (float)q[base + (size_t)t * DMODEL + lane + 32];
        __threadfence_block();   // LDS in-order per wave; fence stops reordering

        float lmax = -1e30f;
        for (int j = lane; j <= t; j += 32) {
            float s = 0.f;
            #pragma unroll
            for (int c = 0; c < HSZ; ++c) s += Qrow[wave][c] * (float)Ks[j][c];
            s *= scale;
            Ps[wave][j] = s;
            lmax = fmaxf(lmax, s);
        }
        #pragma unroll
        for (int off = 16; off; off >>= 1) lmax = fmaxf(lmax, __shfl_xor(lmax, off, 32));
        float lsum = 0.f;
        for (int j = lane; j <= t; j += 32) {
            float e = __expf(Ps[wave][j] - lmax);
            Ps[wave][j] = e;
            lsum += e;
        }
        #pragma unroll
        for (int off = 16; off; off >>= 1) lsum += __shfl_xor(lsum, off, 32);
        const float inv = 1.f / lsum;
        __threadfence_block();

        float o0 = 0.f, o1 = 0.f;
        for (int j = 0; j <= t; ++j) {
            float p = Ps[wave][j];
            o0 += p * (float)Vs[j][lane];
            o1 += p * (float)Vs[j][lane + 32];
        }
        out[base + (size_t)t * DMODEL + lane]      = o0 * inv;
        out[base + (size_t)t * DMODEL + lane + 32] = o1 * inv;
        __threadfence_block();
    }
}

// ---------------------------------------------------------------- loss
__global__ __launch_bounds__(256) void nll_kernel(
    const float* __restrict__ logits, const int* __restrict__ targets,
    float* __restrict__ nll) {
    __shared__ float red[256];
    const int row = blockIdx.x, tid = threadIdx.x;
    const float* lr = logits + (size_t)row * VOCAB;
    float m = -1e30f;
    for (int i = tid; i < VOCAB; i += 256) m = fmaxf(m, lr[i]);
    red[tid] = m; __syncthreads();
    for (int off = 128; off; off >>= 1) {
        if (tid < off) red[tid] = fmaxf(red[tid], red[tid + off]);
        __syncthreads();
    }
    const float rowmax = red[0];
    __syncthreads();
    float s = 0.f;
    for (int i = tid; i < VOCAB; i += 256) s += __expf(lr[i] - rowmax);
    red[tid] = s; __syncthreads();
    for (int off = 128; off; off >>= 1) {
        if (tid < off) red[tid] += red[tid + off];
        __syncthreads();
    }
    if (tid == 0)
        nll[row] = (logf(red[0]) + rowmax) - lr[targets[row]];
}

__global__ __launch_bounds__(256) void mean_kernel(const float* __restrict__ nll,
                                                   float* __restrict__ loss) {
    __shared__ float red[256];
    float s = 0.f;
    for (int i = threadIdx.x; i < MROWS; i += 256) s += nll[i];
    red[threadIdx.x] = s; __syncthreads();
    for (int off = 128; off; off >>= 1) {
        if (threadIdx.x < off) red[threadIdx.x] += red[threadIdx.x + off];
        __syncthreads();
    }
    if (threadIdx.x == 0) loss[0] = red[0] * (1.f / MROWS);
}

// ---------------------------------------------------------------- launch
extern "C" void kernel_launch(void* const* d_in, const int* in_sizes, int n_in,
                              void* d_out, int out_size, void* d_ws, size_t ws_size,
                              hipStream_t stream) {
    (void)in_sizes; (void)n_in; (void)out_size; (void)ws_size;
    const int*   idx     = (const int*)  d_in[0];
    const int*   targets = (const int*)  d_in[1];
    const float* tok_emb = (const float*)d_in[2];
    const float* pos_emb = (const float*)d_in[3];
    const float* Wq      = (const float*)d_in[4];
    const float* Wk      = (const float*)d_in[5];
    const float* Wv      = (const float*)d_in[6];
    const float* Wproj   = (const float*)d_in[7];
    const float* bproj   = (const float*)d_in[8];
    const float* ln1_w   = (const float*)d_in[9];
    const float* ln1_b   = (const float*)d_in[10];
    const float* ln2_w   = (const float*)d_in[11];
    const float* ln2_b   = (const float*)d_in[12];
    const float* W1      = (const float*)d_in[13];
    const float* b1      = (const float*)d_in[14];
    const float* W2      = (const float*)d_in[15];
    const float* b2      = (const float*)d_in[16];
    const float* lnf_w   = (const float*)d_in[17];
    const float* lnf_b   = (const float*)d_in[18];
    const float* Whead   = (const float*)d_in[19];
    const float* bhead   = (const float*)d_in[20];

    float* logits = (float*)d_out;
    float* loss   = logits + (size_t)MROWS * VOCAB;

    // Workspace (~71 MB): f32 x/h/attn/ff1 + f16 q/k/v
    float* ws = (float*)d_ws;
    const size_t MD = (size_t)MROWS * DMODEL;
    float* x     = ws;
    float* hbuf  = x + MD;
    float* attno = hbuf + MD;
    float* ff1   = attno + MD;
    float* nll   = ff1 + (size_t)MROWS * DFFN;
    _Float16* q16 = (_Float16*)(nll + MROWS);
    _Float16* k16 = q16 + MD;
    _Float16* v16 = k16 + MD;

    const dim3 blk(256);
    const dim3 gD(DMODEL / 128, MROWS / 128);   // N=1024 GEMMs
    const dim3 gF(DFFN / 128,   MROWS / 128);   // N=4096 GEMM
    const dim3 gV((VOCAB + 127) / 128, MROWS / 128);

    embed_kernel<<<MROWS, blk, 0, stream>>>(idx, tok_emb, pos_emb, x);

    for (int l = 0; l < NLAYER; ++l) {
        const float* Wq_l    = Wq    + (size_t)l * DMODEL * DMODEL;
        const float* Wk_l    = Wk    + (size_t)l * DMODEL * DMODEL;
        const float* Wv_l    = Wv    + (size_t)l * DMODEL * DMODEL;
        const float* Wproj_l = Wproj + (size_t)l * DMODEL * DMODEL;
        const float* bproj_l = bproj + (size_t)l * DMODEL;
        const float* W1_l    = W1    + (size_t)l * DMODEL * DFFN;
        const float* b1_l    = b1    + (size_t)l * DFFN;
        const float* W2_l    = W2    + (size_t)l * DFFN * DMODEL;
        const float* b2_l    = b2    + (size_t)l * DMODEL;

        layernorm_kernel<<<MROWS, blk, 0, stream>>>(x, ln1_w + (size_t)l * DMODEL,
                                                    ln1_b + (size_t)l * DMODEL, hbuf);
        gemm_wmma_kernel<1, false, false, false, true><<<gD, blk, 0, stream>>>(
            hbuf, Wq_l, nullptr, nullptr, (void*)q16, DMODEL, DMODEL);
        gemm_wmma_kernel<1, false, false, false, true><<<gD, blk, 0, stream>>>(
            hbuf, Wk_l, nullptr, nullptr, (void*)k16, DMODEL, DMODEL);
        gemm_wmma_kernel<1, false, false, false, true><<<gD, blk, 0, stream>>>(
            hbuf, Wv_l, nullptr, nullptr, (void*)v16, DMODEL, DMODEL);

        attention_kernel<<<BATCH * NHEAD, blk, 0, stream>>>(q16, k16, v16, attno);

        // x = x + attno @ Wproj + bproj (in-place residual: each thread reads
        // its own resid element before writing it)
        gemm_wmma_kernel<0, true, false, true, false><<<gD, blk, 0, stream>>>(
            attno, Wproj_l, bproj_l, x, x, DMODEL, DMODEL);

        layernorm_kernel<<<MROWS, blk, 0, stream>>>(x, ln2_w + (size_t)l * DMODEL,
                                                    ln2_b + (size_t)l * DMODEL, hbuf);
        // ff1 = relu(h @ W1 + b1)
        gemm_wmma_kernel<0, true, true, false, false><<<gF, blk, 0, stream>>>(
            hbuf, W1_l, b1_l, nullptr, ff1, DFFN, DMODEL);
        // x = x + ff1 @ W2 + b2
        gemm_wmma_kernel<0, true, false, true, false><<<gD, blk, 0, stream>>>(
            ff1, W2_l, b2_l, x, x, DMODEL, DFFN);
    }

    layernorm_kernel<<<MROWS, blk, 0, stream>>>(x, lnf_w, lnf_b, hbuf);
    // logits = h @ Whead + bhead (straight into d_out)
    gemm_wmma_kernel<0, true, false, false, false><<<gV, blk, 0, stream>>>(
        hbuf, Whead, bhead, nullptr, logits, VOCAB, DMODEL);

    nll_kernel<<<MROWS, blk, 0, stream>>>(logits, targets, nll);
    mean_kernel<<<1, blk, 0, stream>>>(nll, loss);
}
